// LSTM_22737556865709
// MI455X (gfx1250) — matrix operand order repaired and checked by reading, not compiled, not run on hardware
//
#include <hip/hip_runtime.h>

#define HH   1024
#define EE   1024
#define BB   64
#define TT   512
#define KTOT 2048   // H + E
#define N4H  4096   // 4 * H
#define NBLK 64     // persistent blocks (H/16), one wave each

typedef __attribute__((ext_vector_type(16))) __bf16 v16bf;
typedef __attribute__((ext_vector_type(8)))  float  v8f;

union Frag {
    uint4          u4[2];
    unsigned short s[16];
    v16bf          v;
};

__device__ __forceinline__ unsigned short f2bf(float f) {
    unsigned int u = __builtin_bit_cast(unsigned int, f);
    unsigned int r = u + 0x7FFFu + ((u >> 16) & 1u);   // round-to-nearest-even
    return (unsigned short)(r >> 16);
}

__device__ __forceinline__ float fast_sigmoid(float x) {
    return 1.0f / (1.0f + __expf(-x));
}
__device__ __forceinline__ float fast_tanh(float x) {
    return 1.0f - 2.0f / (__expf(2.0f * x) + 1.0f);
}

__device__ __forceinline__ void cvt_x_frag(const float* __restrict__ xp, Frag& a) {
    float4 f0 = *(const float4*)(xp);
    float4 f1 = *(const float4*)(xp + 4);
    float4 f2 = *(const float4*)(xp + 16);
    float4 f3 = *(const float4*)(xp + 20);
    a.s[0]  = f2bf(f0.x); a.s[1]  = f2bf(f0.y); a.s[2]  = f2bf(f0.z); a.s[3]  = f2bf(f0.w);
    a.s[4]  = f2bf(f1.x); a.s[5]  = f2bf(f1.y); a.s[6]  = f2bf(f1.z); a.s[7]  = f2bf(f1.w);
    a.s[8]  = f2bf(f2.x); a.s[9]  = f2bf(f2.y); a.s[10] = f2bf(f2.z); a.s[11] = f2bf(f2.w);
    a.s[12] = f2bf(f3.x); a.s[13] = f2bf(f3.y); a.s[14] = f2bf(f3.z); a.s[15] = f2bf(f3.w);
}

// w[k][n] (f32, 2048x4096) -> wT[n][k] (bf16, 4096x2048), write-coalesced.
__global__ void __launch_bounds__(256)
lstm_wconv(const float* __restrict__ w, unsigned short* __restrict__ wT) {
    int idx = blockIdx.x * 256 + threadIdx.x;      // over N4H * KTOT
    int n = idx / KTOT;
    int k = idx - n * KTOT;
    wT[(size_t)n * KTOT + k] = f2bf(w[(size_t)k * N4H + n]);
}

// zero h0 (bf16, B*H) and the step-barrier counter
__global__ void __launch_bounds__(256)
lstm_zinit(unsigned short* __restrict__ h0, unsigned int* __restrict__ ctr) {
    int i = blockIdx.x * 256 + threadIdx.x;
    h0[i] = 0;
    if (i == 0) *ctr = 0u;
}

// Parallel GEMM over all timesteps: xpart[t*B+b, :] = x @ Wx + bias.
// M = T*B = 32768, K = E = 1024, N = 4H = 4096.
// grid = (N/64, M/(16*8)); block = 256 (8 waves); wave: 16 rows x 64 cols.
__global__ void __launch_bounds__(256)
lstm_xgemm(const float* __restrict__ embeds,       // [T*B, E] f32
           const unsigned short* __restrict__ wT,  // [4H, KTOT] bf16 (x-rows at +HH)
           const float* __restrict__ bias,         // [4H]
           float* __restrict__ xpart)              // [T*B, 4H] f32
{
    const int lane  = threadIdx.x & 31;
    const int wave  = threadIdx.x >> 5;            // 0..7
    const int hi    = lane >> 4;
    const int l15   = lane & 15;
    const int mtile = blockIdx.y * 8 + wave;       // 0..2047
    const int nbase = blockIdx.x * 64;

    const float* xrow = embeds + (size_t)((mtile << 4) + l15) * EE;
    const unsigned short* wg[4];
#pragma unroll
    for (int t = 0; t < 4; ++t)
        wg[t] = wT + (size_t)(nbase + t * 16 + l15) * KTOT + HH;  // x-half of K

    v8f acc[4] = {};
    for (int kt = 0; kt < EE / 32; ++kt) {
        const int ks = kt << 5;
        const int ca = ks + (hi ? 8 : 0);
        const int kb = ks + (hi ? 8 : 0);
        Frag a;
        cvt_x_frag(xrow + ca, a);
#pragma unroll
        for (int t = 0; t < 4; ++t) {
            Frag b;
            b.u4[0] = *(const uint4*)(wg[t] + kb);
            b.u4[1] = *(const uint4*)(wg[t] + kb + 16);
            acc[t] = __builtin_amdgcn_wmma_f32_16x16x32_bf16(false, a.v, false, b.v, (short)0, acc[t], false, false);
        }
    }
#pragma unroll
    for (int t = 0; t < 4; ++t) {
#pragma unroll
        for (int r = 0; r < 8; ++r) {
            const int m = (mtile << 4) + (hi << 3) + r;
            const int n = nbase + t * 16 + l15;
            xpart[(size_t)m * N4H + n] = acc[t][r] + bias[n];
        }
    }
}

// Persistent recurrence: ONE launch runs all T=512 steps.
// grid = 64 single-wave blocks; each wave owns all 64 batch rows (4 M-tiles)
// x 16 cols of each of 4 gates. The cell state c stays in registers for the
// whole sequence; steps are separated by a device-scope monotonic-counter
// barrier (all 64 blocks are trivially co-resident).
template <bool USE_XPART>
__global__ void __launch_bounds__(32)
lstm_persist(const float* __restrict__ embeds,       // [T, B, E] f32 (fallback)
             const float* __restrict__ xpart,        // [T*B, 4H] f32 (xpart path)
             const unsigned short* __restrict__ wT,  // [4H, KTOT] bf16
             const float* __restrict__ bias,         // [4H] (fallback only)
             unsigned short* __restrict__ hb0,       // [B, H] bf16 (holds h(0)=0)
             unsigned short* __restrict__ hb1,       // [B, H] bf16
             float* __restrict__ out,                // [T, B, H] f32
             unsigned int* __restrict__ bar_ctr)
{
    const int lane = threadIdx.x & 31;
    const int hi   = lane >> 4;
    const int l15  = lane & 15;
    const int n0   = blockIdx.x << 4;               // col group within H
    const int ncol = n0 + l15;
    const int jcol = n0 + l15;

    const unsigned short* wg[4];
#pragma unroll
    for (int g = 0; g < 4; ++g)
        wg[g] = wT + (size_t)(ncol + g * HH) * KTOT;

    float bg = 0.f, bi = 0.f, bf = 0.f, bo = 0.f;
    if (!USE_XPART) {
        bg = bias[jcol];
        bi = bias[HH + jcol];
        bf = bias[2 * HH + jcol];
        bo = bias[3 * HH + jcol];
    }

    float c_reg[4][8];                              // cell state, in registers
#pragma unroll
    for (int mt = 0; mt < 4; ++mt)
#pragma unroll
        for (int r = 0; r < 8; ++r) c_reg[mt][r] = 0.f;

    for (int t = 0; t < TT; ++t) {
        const unsigned short* h_in  = (t & 1) ? hb1 : hb0;
        unsigned short*       h_out = (t & 1) ? hb0 : hb1;

        v8f acc[4][4] = {};                         // [gate][mtile]

        // ---- K in [0, H): recurrent half, pure b128 loads + WMMA ----
        for (int kt = 0; kt < HH / 32; ++kt) {
            const int ks = kt << 5;
            const int ca = ks + (hi ? 8 : 0);
            const int kb = ks + (hi ? 8 : 0);
            Frag a[4];
#pragma unroll
            for (int mt = 0; mt < 4; ++mt) {
                const unsigned short* hrow = h_in + (size_t)((mt << 4) + l15) * HH;
                a[mt].u4[0] = *(const uint4*)(hrow + ca);
                a[mt].u4[1] = *(const uint4*)(hrow + ca + 16);
            }
#pragma unroll
            for (int g = 0; g < 4; ++g) {
                Frag b;
                b.u4[0] = *(const uint4*)(wg[g] + kb);
                b.u4[1] = *(const uint4*)(wg[g] + kb + 16);
#pragma unroll
                for (int mt = 0; mt < 4; ++mt)
                    acc[g][mt] = __builtin_amdgcn_wmma_f32_16x16x32_bf16(
                        false, a[mt].v, false, b.v, (short)0, acc[g][mt], false, false);
            }
        }

        // ---- fallback: K in [H, H+E) fused, x converted on the fly ----
        if (!USE_XPART) {
            const float* x_t = embeds + (size_t)t * BB * EE;
            for (int kt = 0; kt < EE / 32; ++kt) {
                const int ks = kt << 5;
                const int ca = ks + (hi ? 8 : 0);
                const int kb = HH + ks + (hi ? 8 : 0);
                Frag a[4];
#pragma unroll
                for (int mt = 0; mt < 4; ++mt)
                    cvt_x_frag(x_t + (size_t)((mt << 4) + l15) * EE + ca, a[mt]);
#pragma unroll
                for (int g = 0; g < 4; ++g) {
                    Frag b;
                    b.u4[0] = *(const uint4*)(wg[g] + kb);
                    b.u4[1] = *(const uint4*)(wg[g] + kb + 16);
#pragma unroll
                    for (int mt = 0; mt < 4; ++mt)
                        acc[g][mt] = __builtin_amdgcn_wmma_f32_16x16x32_bf16(
                            false, a[mt].v, false, b.v, (short)0, acc[g][mt], false, false);
                }
            }
        }

        // ---- epilogue: gates + cell/hidden update (c in registers) ----
        float* out_t = out + (size_t)t * BB * HH;
        const float* xp_t = xpart + (size_t)t * BB * N4H;
#pragma unroll
        for (int mt = 0; mt < 4; ++mt) {
#pragma unroll
            for (int r = 0; r < 8; ++r) {
                const int m = (mt << 4) + (hi << 3) + r;
                const size_t off = (size_t)m * HH + jcol;
                float pg, pi, pf, po;
                if (USE_XPART) {
                    const float* xp = xp_t + (size_t)m * N4H + jcol;
                    pg = xp[0];
                    pi = xp[HH];
                    pf = xp[2 * HH];
                    po = xp[3 * HH];
                } else {
                    pg = bg; pi = bi; pf = bf; po = bo;
                }
                float g = fast_tanh(acc[0][mt][r] + pg);
                float i = fast_sigmoid(acc[1][mt][r] + pi);
                float f = fast_sigmoid(acc[2][mt][r] + pf);
                float o = fast_sigmoid(acc[3][mt][r] + po);
                float cn = g * i + c_reg[mt][r] * f;
                c_reg[mt][r] = cn;
                float hn = fast_tanh(cn) * o;
                out_t[off] = hn;
                h_out[off] = f2bf(hn);
            }
        }

        // ---- inter-step grid barrier (monotonic counter) ----
        __threadfence();                            // make h_out visible (device scope)
        if (lane == 0) atomicAdd(bar_ctr, 1u);
        const unsigned int target = (unsigned int)NBLK * (unsigned int)(t + 1);
        while (__hip_atomic_load(bar_ctr, __ATOMIC_RELAXED, __HIP_MEMORY_SCOPE_AGENT) < target) {
            __builtin_amdgcn_s_sleep(2);
        }
        __threadfence();                            // acquire h from other blocks
    }
}

extern "C" void kernel_launch(void* const* d_in, const int* in_sizes, int n_in,
                              void* d_out, int out_size, void* d_ws, size_t ws_size,
                              hipStream_t stream) {
    const float* embeds = (const float*)d_in[0];   // [T, B, E] f32
    const float* w      = (const float*)d_in[1];   // [KTOT, 4H] f32
    const float* bias   = (const float*)d_in[2];   // [4H] f32
    float* out = (float*)d_out;                    // [T, B, H] f32

    // workspace layout:
    //   wT    : 4096*2048 bf16          = 16 MB
    //   h0,h1 : 2 * 64*1024 bf16        = 256 KB
    //   ctr   : barrier counter          (256 B slot)
    //   xpart : 32768*4096 f32          = 512 MB   (optional, if ws permits)
    char* ws = (char*)d_ws;
    unsigned short* wT = (unsigned short*)ws;
    unsigned short* h0 = (unsigned short*)(ws + (size_t)N4H * KTOT * 2);
    unsigned short* h1 = h0 + BB * HH;
    unsigned int* ctr = (unsigned int*)(h1 + BB * HH);
    char* head_end = (char*)ctr + 256;
    float* xpart = (float*)head_end;

    const size_t head_bytes  = (size_t)(head_end - ws);
    const size_t xpart_bytes = (size_t)TT * BB * N4H * sizeof(float);
    const bool use_xpart = (ws_size >= head_bytes + xpart_bytes);

    lstm_wconv<<<(N4H * KTOT) / 256, 256, 0, stream>>>(w, wT);
    lstm_zinit<<<(BB * HH) / 256, 256, 0, stream>>>(h0, ctr);

    if (use_xpart) {
        // all-timestep input GEMM: M=T*B=32768 -> grid.y = 32768/16/8 = 256
        lstm_xgemm<<<dim3(N4H / 64, (TT * BB) / (16 * 8)), 256, 0, stream>>>(
            embeds, wT, bias, xpart);
        lstm_persist<true><<<NBLK, 32, 0, stream>>>(
            embeds, xpart, wT, bias, h0, h1, out, ctr);
    } else {
        lstm_persist<false><<<NBLK, 32, 0, stream>>>(
            embeds, xpart, wT, bias, h0, h1, out, ctr);
    }
}
